// Net_36017595744901
// MI455X (gfx1250) — compile-verified, matrix operand order checked by reading, not run on hardware
//
#include <hip/hip_runtime.h>
#include <hip/hip_bf16.h>
#include <math.h>

typedef __attribute__((ext_vector_type(16))) _Float16 v16h;
typedef __attribute__((ext_vector_type(8)))  _Float16 v8h;
typedef __attribute__((ext_vector_type(8)))  float    v8f;

#define N_NODES 50000
#define N_EDGES 1600000
#define ET      (N_EDGES + N_NODES)   // edges incl. self loops
#define F_IN    1433
#define KPAD    1440                  // 45 * 32
#define KMAIN   1408                  // 44 * 32 (fully in-bounds k-steps)
#define NTILES  3125                  // 50000 / 16 row tiles
#define NH1     8
#define C1      8
#define D1      64                    // NH1*C1
#define C2      7
#define SLOPE   0.2f
#define ENC_NEG_INF 0x007FFFFFu       // enc(-inf)

// ---- monotonic float <-> uint encoding so float-max == uint-max ----
__device__ __forceinline__ unsigned fenc(float f) {
    unsigned u = __float_as_uint(f);
    return u ^ ((u >> 31) ? 0xFFFFFFFFu : 0x80000000u);
}
__device__ __forceinline__ float fdec(unsigned u) {
    return __uint_as_float((u >> 31) ? (u ^ 0x80000000u) : ~u);
}

// ======================= prep: W1^T as f16 [64][KPAD] =======================
__global__ void k_prep_w1t(const float* __restrict__ W1, _Float16* __restrict__ w1t) {
    int tid = blockIdx.x * blockDim.x + threadIdx.x;   // over 64*KPAD
    if (tid >= D1 * KPAD) return;
    int n = tid / KPAD, k = tid % KPAD;
    float v = (k < F_IN) ? W1[k * D1 + n] : 0.0f;
    w1t[n * KPAD + k] = (_Float16)v;
}

// ======================= init scratch every call ============================
__global__ void k_init(float* agg1, float* den1, unsigned* m1,
                       float* agg2, float* den2, unsigned* m2) {
    int tid = blockIdx.x * blockDim.x + threadIdx.x;   // over N*64
    if (tid < N_NODES * D1) agg1[tid] = 0.0f;
    if (tid < N_NODES * NH1) { den1[tid] = 0.0f; m1[tid] = ENC_NEG_INF; }
    if (tid < N_NODES * C2)  agg2[tid] = 0.0f;
    if (tid < N_NODES)       { den2[tid] = 0.0f; m2[tid] = ENC_NEG_INF; }
}

// ======================= layer-1 GEMM via WMMA ==============================
// grid.x = ceil(3125/4) blocks of 128 threads = 4 waves.
// Each wave owns one 16-row tile and computes all 4 column tiles (64 cols),
// so the X fragment is loaded once and reused for 4 WMMAs.
__global__ void k_gemm1(const float* __restrict__ X,
                        const _Float16* __restrict__ w1t,
                        float* __restrict__ h1) {
    const int lane = threadIdx.x & 31;
    const int wave = threadIdx.x >> 5;               // 0..3
    const int tile = blockIdx.x * 4 + wave;          // row tile (wave-uniform)
    if (tile >= NTILES) return;                      // wave-uniform: EXEC stays full

    const int m   = lane & 15;
    const int kb  = (lane < 16) ? 0 : 8;             // k sub-offset for this lane half
    const int row = tile * 16 + m;

    const float*    xrow  = X + (size_t)row * F_IN;
    const _Float16* bbase = w1t + (size_t)m * KPAD;  // column tile t adds t*16*KPAD

    v8f acc0 = {}, acc1 = {}, acc2 = {}, acc3 = {};

    // ---------------- main loop: fully in-bounds, no guards ----------------
    for (int ks = 0; ks < KMAIN; ks += 32) {
        const int k0 = ks + kb;
        v16h a;
        #pragma unroll
        for (int i = 0; i < 8; ++i) a[i]     = (_Float16)xrow[k0 + i];
        #pragma unroll
        for (int i = 0; i < 8; ++i) a[8 + i] = (_Float16)xrow[k0 + 16 + i];

        #pragma unroll
        for (int t = 0; t < 4; ++t) {
            const _Float16* bp = bbase + (size_t)t * 16 * KPAD + k0;
            v8h blo = *(const v8h*)(bp);
            v8h bhi = *(const v8h*)(bp + 16);
            v16h b;
            #pragma unroll
            for (int i = 0; i < 8; ++i) { b[i] = blo[i]; b[8 + i] = bhi[i]; }
            v8f* pc = (t == 0) ? &acc0 : (t == 1) ? &acc1 : (t == 2) ? &acc2 : &acc3;
            *pc = __builtin_amdgcn_wmma_f32_16x16x32_f16(
                      false, a, false, b, (short)0, *pc, false, false);
        }
    }

    // ---------------- peeled tail: ks = KMAIN, guard only chunk 2 ----------
    {
        const int k0 = KMAIN + kb;
        v16h a;
        #pragma unroll
        for (int i = 0; i < 8; ++i) a[i] = (_Float16)xrow[k0 + i];   // max 1423 < F_IN
        #pragma unroll
        for (int i = 0; i < 8; ++i) {
            int ka = k0 + 16 + i;                                    // up to 1439
            a[8 + i] = (_Float16)((ka < F_IN) ? xrow[ka] : 0.0f);
        }
        #pragma unroll
        for (int t = 0; t < 4; ++t) {
            const _Float16* bp = bbase + (size_t)t * 16 * KPAD + k0;
            v8h blo = *(const v8h*)(bp);
            v8h bhi = *(const v8h*)(bp + 16);
            v16h b;
            #pragma unroll
            for (int i = 0; i < 8; ++i) { b[i] = blo[i]; b[8 + i] = bhi[i]; }
            v8f* pc = (t == 0) ? &acc0 : (t == 1) ? &acc1 : (t == 2) ? &acc2 : &acc3;
            *pc = __builtin_amdgcn_wmma_f32_16x16x32_f16(
                      false, a, false, b, (short)0, *pc, false, false);
        }
    }

    // ---- C/D layout: VGPR r -> M = r (lanes 0-15) / 8+r (lanes 16-31) ----
    const int rbase = tile * 16 + ((lane >> 4) & 1) * 8;
    #pragma unroll
    for (int t = 0; t < 4; ++t) {
        const v8f& c = (t == 0) ? acc0 : (t == 1) ? acc1 : (t == 2) ? acc2 : acc3;
        const int col = t * 16 + (lane & 15);
        #pragma unroll
        for (int r = 0; r < 8; ++r)
            h1[(size_t)(rbase + r) * D1 + col] = c[r];
    }
}

// ======================= per-node attention logits (layer 1) ================
__global__ void k_alpha1(const float* __restrict__ h1,
                         const float* __restrict__ a_src,
                         const float* __restrict__ a_dst,
                         float* __restrict__ as1, float* __restrict__ ad1) {
    int tid = blockIdx.x * blockDim.x + threadIdx.x;   // over N*8
    if (tid >= N_NODES * NH1) return;
    int n = tid >> 3, h = tid & 7;
    const float* hr = h1 + (size_t)n * D1 + h * C1;
    float s = 0.f, d = 0.f;
    #pragma unroll
    for (int c = 0; c < C1; ++c) {
        s += hr[c] * a_src[h * C1 + c];
        d += hr[c] * a_dst[h * C1 + c];
    }
    as1[tid] = s; ad1[tid] = d;
}

__device__ __forceinline__ void edge_sd(int e, const int* src, const int* dst,
                                        int& s, int& d) {
    if (e < N_EDGES) { s = src[e]; d = dst[e]; }
    else             { s = d = e - N_EDGES; }   // self loop
}

// ======================= layer-1 edge passes ================================
__global__ void k_e1_max(const int* src, const int* dst,
                         const float* as1, const float* ad1, unsigned* m1) {
    int tid = blockIdx.x * blockDim.x + threadIdx.x;   // over ET*8
    if (tid >= ET * NH1) return;
    int e = tid >> 3, h = tid & 7, s, d;
    edge_sd(e, src, dst, s, d);
    float a = as1[s * NH1 + h] + ad1[d * NH1 + h];
    float ev = (a > 0.f) ? a : SLOPE * a;
    atomicMax(m1 + d * NH1 + h, fenc(ev));
}

__global__ void k_e1_sum(const int* src, const int* dst,
                         const float* as1, const float* ad1,
                         const unsigned* m1, float* den1) {
    int tid = blockIdx.x * blockDim.x + threadIdx.x;
    if (tid >= ET * NH1) return;
    int e = tid >> 3, h = tid & 7, s, d;
    edge_sd(e, src, dst, s, d);
    float a = as1[s * NH1 + h] + ad1[d * NH1 + h];
    float ev = (a > 0.f) ? a : SLOPE * a;
    atomicAdd(den1 + d * NH1 + h, expf(ev - fdec(m1[d * NH1 + h])));
}

__global__ void k_e1_agg(const int* src, const int* dst,
                         const float* as1, const float* ad1,
                         const unsigned* m1, const float* den1,
                         const float* h1, float* agg1) {
    int tid = blockIdx.x * blockDim.x + threadIdx.x;
    if (tid >= ET * NH1) return;
    int e = tid >> 3, h = tid & 7, s, d;
    edge_sd(e, src, dst, s, d);
    float a = as1[s * NH1 + h] + ad1[d * NH1 + h];
    float ev = (a > 0.f) ? a : SLOPE * a;
    float alpha = expf(ev - fdec(m1[d * NH1 + h])) / (den1[d * NH1 + h] + 1e-16f);
    const float* hs = h1 + (size_t)s * D1 + h * C1;
    float* od = agg1 + (size_t)d * D1 + h * C1;
    #pragma unroll
    for (int c = 0; c < C1; ++c) atomicAdd(od + c, hs[c] * alpha);
}

// ======================= bias + ELU (writes h1 in place) ====================
__global__ void k_fin1(const float* __restrict__ agg1, const float* __restrict__ b1,
                       float* __restrict__ h1) {
    int tid = blockIdx.x * blockDim.x + threadIdx.x;   // over N*64
    if (tid >= N_NODES * D1) return;
    float v = agg1[tid] + b1[tid % D1];
    h1[tid] = (v > 0.f) ? v : (expf(v) - 1.0f);        // ELU
}

// ======================= layer-2 GEMM + attention logits ====================
__global__ void k_gemm2(const float* __restrict__ h1, const float* __restrict__ W2,
                        const float* __restrict__ a_src2, const float* __restrict__ a_dst2,
                        float* __restrict__ z2, float* __restrict__ as2,
                        float* __restrict__ ad2) {
    int n = blockIdx.x * blockDim.x + threadIdx.x;     // over N
    if (n >= N_NODES) return;
    const float* hr = h1 + (size_t)n * D1;
    float acc[C2];
    #pragma unroll
    for (int c = 0; c < C2; ++c) acc[c] = 0.f;
    for (int j = 0; j < D1; ++j) {
        float hv = hr[j];
        #pragma unroll
        for (int c = 0; c < C2; ++c) acc[c] += hv * W2[j * C2 + c];
    }
    float s = 0.f, d = 0.f;
    #pragma unroll
    for (int c = 0; c < C2; ++c) {
        z2[n * C2 + c] = acc[c];
        s += acc[c] * a_src2[c];
        d += acc[c] * a_dst2[c];
    }
    as2[n] = s; ad2[n] = d;
}

// ======================= layer-2 edge passes (1 head) =======================
__global__ void k_e2_max(const int* src, const int* dst,
                         const float* as2, const float* ad2, unsigned* m2) {
    int e = blockIdx.x * blockDim.x + threadIdx.x;
    if (e >= ET) return;
    int s, d; edge_sd(e, src, dst, s, d);
    float a = as2[s] + ad2[d];
    float ev = (a > 0.f) ? a : SLOPE * a;
    atomicMax(m2 + d, fenc(ev));
}

__global__ void k_e2_sum(const int* src, const int* dst,
                         const float* as2, const float* ad2,
                         const unsigned* m2, float* den2) {
    int e = blockIdx.x * blockDim.x + threadIdx.x;
    if (e >= ET) return;
    int s, d; edge_sd(e, src, dst, s, d);
    float a = as2[s] + ad2[d];
    float ev = (a > 0.f) ? a : SLOPE * a;
    atomicAdd(den2 + d, expf(ev - fdec(m2[d])));
}

__global__ void k_e2_agg(const int* src, const int* dst,
                         const float* as2, const float* ad2,
                         const unsigned* m2, const float* den2,
                         const float* z2, float* agg2) {
    int e = blockIdx.x * blockDim.x + threadIdx.x;
    if (e >= ET) return;
    int s, d; edge_sd(e, src, dst, s, d);
    float a = as2[s] + ad2[d];
    float ev = (a > 0.f) ? a : SLOPE * a;
    float alpha = expf(ev - fdec(m2[d])) / (den2[d] + 1e-16f);
    #pragma unroll
    for (int c = 0; c < C2; ++c)
        atomicAdd(agg2 + d * C2 + c, z2[s * C2 + c] * alpha);
}

// ======================= bias + log_softmax =================================
__global__ void k_fin2(const float* __restrict__ agg2, const float* __restrict__ b2,
                       float* __restrict__ out) {
    int n = blockIdx.x * blockDim.x + threadIdx.x;
    if (n >= N_NODES) return;
    float o[C2], mx = -1e30f;
    #pragma unroll
    for (int c = 0; c < C2; ++c) {
        o[c] = agg2[n * C2 + c] + b2[c];
        mx = fmaxf(mx, o[c]);
    }
    float se = 0.f;
    #pragma unroll
    for (int c = 0; c < C2; ++c) se += expf(o[c] - mx);
    float lse = mx + logf(se);
    #pragma unroll
    for (int c = 0; c < C2; ++c) out[n * C2 + c] = o[c] - lse;
}

// ============================================================================
static inline int cdiv(long long a, int b) { return (int)((a + b - 1) / b); }

extern "C" void kernel_launch(void* const* d_in, const int* in_sizes, int n_in,
                              void* d_out, int out_size, void* d_ws, size_t ws_size,
                              hipStream_t stream) {
    const float* x      = (const float*)d_in[0];
    const int*   eidx   = (const int*)d_in[1];
    const float* W1     = (const float*)d_in[2];
    const float* asrc1  = (const float*)d_in[3];
    const float* adst1  = (const float*)d_in[4];
    const float* b1     = (const float*)d_in[5];
    const float* W2     = (const float*)d_in[6];
    const float* asrc2  = (const float*)d_in[7];
    const float* adst2  = (const float*)d_in[8];
    const float* b2     = (const float*)d_in[9];
    const int* src = eidx;
    const int* dst = eidx + N_EDGES;
    float* out = (float*)d_out;

    // ---- workspace carve-up (all offsets 64B aligned) ----
    char* ws = (char*)d_ws;
    size_t off = 0;
    _Float16* w1t = (_Float16*)(ws + off); off += (size_t)D1 * KPAD * 2;          // 184320
    float*    h1  = (float*)(ws + off);    off += (size_t)N_NODES * D1 * 4;       // 12.8 MB
    float*    as1 = (float*)(ws + off);    off += (size_t)N_NODES * NH1 * 4;
    float*    ad1 = (float*)(ws + off);    off += (size_t)N_NODES * NH1 * 4;
    unsigned* m1  = (unsigned*)(ws + off); off += (size_t)N_NODES * NH1 * 4;
    float*    den1= (float*)(ws + off);    off += (size_t)N_NODES * NH1 * 4;
    float*    agg1= (float*)(ws + off);    off += (size_t)N_NODES * D1 * 4;       // 12.8 MB
    float*    z2  = (float*)(ws + off);    off += (size_t)N_NODES * C2 * 4;
    float*    as2 = (float*)(ws + off);    off += (size_t)N_NODES * 4;
    float*    ad2 = (float*)(ws + off);    off += (size_t)N_NODES * 4;
    unsigned* m2  = (unsigned*)(ws + off); off += (size_t)N_NODES * 4;
    float*    den2= (float*)(ws + off);    off += (size_t)N_NODES * 4;
    float*    agg2= (float*)(ws + off);    off += (size_t)N_NODES * C2 * 4;

    const int B = 256;
    // prep + init
    k_prep_w1t<<<cdiv((long long)D1 * KPAD, B), B, 0, stream>>>(W1, w1t);
    k_init<<<cdiv((long long)N_NODES * D1, B), B, 0, stream>>>(agg1, den1, m1, agg2, den2, m2);
    // layer 1
    k_gemm1<<<cdiv(NTILES, 4), 128, 0, stream>>>(x, w1t, h1);
    k_alpha1<<<cdiv((long long)N_NODES * NH1, B), B, 0, stream>>>(h1, asrc1, adst1, as1, ad1);
    k_e1_max<<<cdiv((long long)ET * NH1, B), B, 0, stream>>>(src, dst, as1, ad1, m1);
    k_e1_sum<<<cdiv((long long)ET * NH1, B), B, 0, stream>>>(src, dst, as1, ad1, m1, den1);
    k_e1_agg<<<cdiv((long long)ET * NH1, B), B, 0, stream>>>(src, dst, as1, ad1, m1, den1, h1, agg1);
    k_fin1<<<cdiv((long long)N_NODES * D1, B), B, 0, stream>>>(agg1, b1, h1);
    // layer 2
    k_gemm2<<<cdiv(N_NODES, B), B, 0, stream>>>(h1, W2, asrc2, adst2, z2, as2, ad2);
    k_e2_max<<<cdiv(ET, B), B, 0, stream>>>(src, dst, as2, ad2, m2);
    k_e2_sum<<<cdiv(ET, B), B, 0, stream>>>(src, dst, as2, ad2, m2, den2);
    k_e2_agg<<<cdiv(ET, B), B, 0, stream>>>(src, dst, as2, ad2, m2, den2, z2, agg2);
    k_fin2<<<cdiv(N_NODES, B), B, 0, stream>>>(agg2, b2, out);
}
